// FasterRCNNVGG16_24455543783979
// MI455X (gfx1250) — matrix operand order repaired and checked by reading, not compiled, not run on hardware
//
#include <hip/hip_runtime.h>

// ---------------------------------------------------------------------------
// CDNA5 (gfx1250) Faster R-CNN VGG16 forward.
// Convs + FCs via v_wmma_f32_16x16x32_bf16 (wave32, 16x16 tiles per wave).
// ---------------------------------------------------------------------------

typedef __attribute__((ext_vector_type(16))) __bf16 v16bf;
typedef __attribute__((ext_vector_type(8)))  __bf16 v8bf;
typedef __attribute__((ext_vector_type(8)))  float  v8f;

// Fragment lane<->element maps (ISA 7.12.2, wave32):
//  A (16Mx32K bf16): lane l row M=l&15; element e -> K=e+((l>>4)<<4) here (contiguous)
//  B (32Kx16N bf16): lane l col N=l&15; element e -> K=e+((l>>4)<<4)   (contiguous here, same as A; ISA table order differs)
//  C/D (16Mx16N f32): lane l col N=l&15; vgpr j row M=j+((l>>4)<<3)

// ---------------------------------------------------------------------------
// Weight repack: OIHW fp32 -> bf16 [tap][Cout][CinPad], zero-padded channels.
// ---------------------------------------------------------------------------
__global__ void repack_w_kernel(const float* __restrict__ w, __bf16* __restrict__ wre,
                                int Cin, int CinPad, int Cout)
{
  int idx = blockIdx.x * blockDim.x + threadIdx.x;
  int total = 9 * Cout * CinPad;
  if (idx >= total) return;
  int ci  = idx % CinPad;
  int co  = (idx / CinPad) % Cout;
  int tap = idx / (CinPad * Cout);
  float v = 0.0f;
  if (ci < Cin) v = w[((size_t)co * Cin + ci) * 9 + tap];  // tap = dy*3+dx
  wre[idx] = (__bf16)v;
}

// ---------------------------------------------------------------------------
// 3x3 conv (pad=1) + bias (+ReLU): 9 shifted WMMA GEMM taps over K=CinPad.
// One wave per block: 16 output channels x 32 consecutive x at one row y
// (two 16x16 C tiles share every A fragment).
// grid = (ceil(W/32), H, Cout/16). CinPad % 32 == 0, weights pre-repacked.
// ---------------------------------------------------------------------------
__global__ __launch_bounds__(32) void conv3x3_wmma(
    const float* __restrict__ in, const __bf16* __restrict__ wre,
    const float* __restrict__ bias, float* __restrict__ out,
    int CinReal, int CinPad, int Cout, int H, int W, int relu)
{
  __shared__ __bf16 patch[3][34][32];  // [row][x (34 used)][cin] -> K contiguous

  const int tx   = blockIdx.x << 5;
  const int y    = blockIdx.y;
  const int co0  = blockIdx.z << 4;
  const int lane = threadIdx.x;

  v8f acc0 = {}, acc1 = {};

  const int mA     = lane & 15;
  const int kbaseA = (lane >> 4) << 4;
  const int nB     = lane & 15;
  const int kbaseB = (lane >> 4) << 4;

  const int nchunk = CinPad >> 5;
  for (int ch = 0; ch < nchunk; ++ch) {
    const int c0 = ch << 5;

    // Stage 32 channels x 3 rows x 34 cols (fp32 -> bf16), x innermost
    // (coalesced global reads); LDS layout keeps cin contiguous for B frags.
    for (int idx = lane; idx < 32 * 3 * 34; idx += 32) {
      int xx = idx % 34;
      int r  = (idx / 34) % 3;
      int c  = idx / (3 * 34);
      int gy = y + r - 1;
      int gx = tx + xx - 1;
      int gc = c0 + c;
      float v = 0.0f;
      if (gc < CinReal && gy >= 0 && gy < H && gx >= 0 && gx < W)
        v = in[((size_t)gc * H + gy) * W + gx];
      patch[r][xx][c] = (__bf16)v;
    }
    __syncthreads();

#pragma unroll
    for (int dy = 0; dy < 3; ++dy) {
#pragma unroll
      for (int dx = 0; dx < 3; ++dx) {
        const int tap = dy * 3 + dx;
        // A fragment: two contiguous 16B chunks of repacked bf16 weights.
        const __bf16* wp =
            wre + ((size_t)(tap * Cout + co0 + mA)) * CinPad + c0 + kbaseA;
        v8bf alo = *(const v8bf*)(wp);
        v8bf ahi = *(const v8bf*)(wp + 8);
        v16bf a = __builtin_shufflevector(alo, ahi, 0, 1, 2, 3, 4, 5, 6, 7,
                                          8, 9, 10, 11, 12, 13, 14, 15);
        // B fragments: one contiguous 32B LDS read each (K contiguous).
        v16bf b0 = *(const v16bf*)&patch[dy][nB + dx][kbaseB];
        v16bf b1 = *(const v16bf*)&patch[dy][nB + 16 + dx][kbaseB];
        acc0 = __builtin_amdgcn_wmma_f32_16x16x32_bf16(
            false, a, false, b0, (short)0, acc0, false, false);
        acc1 = __builtin_amdgcn_wmma_f32_16x16x32_bf16(
            false, a, false, b1, (short)0, acc1, false, false);
      }
    }
    __syncthreads();
  }

  // Epilogue: bias + ReLU, bounds-checked stores for both pixel tiles.
  const int x0 = tx + (lane & 15);
  const int x1 = x0 + 16;
#pragma unroll
  for (int j = 0; j < 8; ++j) {
    int co = co0 + j + ((lane >> 4) << 3);
    float b = bias[co];
    if (x0 < W) {
      float v = acc0[j] + b;
      if (relu) v = fmaxf(v, 0.0f);
      out[((size_t)co * H + y) * W + x0] = v;
    }
    if (x1 < W) {
      float v = acc1[j] + b;
      if (relu) v = fmaxf(v, 0.0f);
      out[((size_t)co * H + y) * W + x1] = v;
    }
  }
}

// ---------------------------------------------------------------------------
// GEMM: out[M,N] = act[M,K] @ W[N,K]^T + bias (+ReLU). One wave per 16x16 tile.
// grid = (ceil(N/16), M/16). K % 32 == 0 (25088/4096). Branchless tail via
// clamped B row (tail columns never stored).
// ---------------------------------------------------------------------------
__global__ __launch_bounds__(32) void gemm_wmma(
    const float* __restrict__ act, const float* __restrict__ Wt,
    const float* __restrict__ bias, float* __restrict__ out,
    int M, int N, int K, int relu)
{
  const int n0   = blockIdx.x << 4;
  const int m0   = blockIdx.y << 4;
  const int lane = threadIdx.x;

  v8f acc = {};
  const int mA     = m0 + (lane & 15);
  const int kbaseA = (lane >> 4) << 4;
  const int rowB   = min(n0 + (lane & 15), N - 1);  // clamp: tail cols discarded
  const int kbaseB = (lane >> 4) << 4;

  const float* pa = act + (size_t)mA * K + kbaseA;
  const float* pb = Wt + (size_t)rowB * K + kbaseB;

  for (int k0 = 0; k0 < K; k0 += 32) {
    float4 a0 = *(const float4*)(pa + k0);
    float4 a1 = *(const float4*)(pa + k0 + 4);
    float4 a2 = *(const float4*)(pa + k0 + 8);
    float4 a3 = *(const float4*)(pa + k0 + 12);
    float4 b0 = *(const float4*)(pb + k0);
    float4 b1 = *(const float4*)(pb + k0 + 4);
    float4 b2 = *(const float4*)(pb + k0 + 8);
    float4 b3 = *(const float4*)(pb + k0 + 12);
    __builtin_prefetch((const void*)(pb + k0 + 64), 0, 0);  // stream weights

    v16bf a, b;
    a[0]=(__bf16)a0.x; a[1]=(__bf16)a0.y; a[2]=(__bf16)a0.z; a[3]=(__bf16)a0.w;
    a[4]=(__bf16)a1.x; a[5]=(__bf16)a1.y; a[6]=(__bf16)a1.z; a[7]=(__bf16)a1.w;
    a[8]=(__bf16)a2.x; a[9]=(__bf16)a2.y; a[10]=(__bf16)a2.z; a[11]=(__bf16)a2.w;
    a[12]=(__bf16)a3.x; a[13]=(__bf16)a3.y; a[14]=(__bf16)a3.z; a[15]=(__bf16)a3.w;
    b[0]=(__bf16)b0.x; b[1]=(__bf16)b0.y; b[2]=(__bf16)b0.z; b[3]=(__bf16)b0.w;
    b[4]=(__bf16)b1.x; b[5]=(__bf16)b1.y; b[6]=(__bf16)b1.z; b[7]=(__bf16)b1.w;
    b[8]=(__bf16)b2.x; b[9]=(__bf16)b2.y; b[10]=(__bf16)b2.z; b[11]=(__bf16)b2.w;
    b[12]=(__bf16)b3.x; b[13]=(__bf16)b3.y; b[14]=(__bf16)b3.z; b[15]=(__bf16)b3.w;

    acc = __builtin_amdgcn_wmma_f32_16x16x32_bf16(
        false, a, false, b, (short)0, acc, false, false);
  }

  const int nc = n0 + (lane & 15);
  if (nc < N) {
#pragma unroll
    for (int j = 0; j < 8; ++j) {
      int m = m0 + j + ((lane >> 4) << 3);
      float v = acc[j] + bias[nc];
      if (relu) v = fmaxf(v, 0.0f);
      out[(size_t)m * N + nc] = v;
    }
  }
}

// ---------------------------------------------------------------------------
// 2x2/2 maxpool (VALID)
// ---------------------------------------------------------------------------
__global__ void maxpool2_kernel(const float* __restrict__ in,
                                float* __restrict__ out, int C, int H, int W)
{
  int Ho = H >> 1, Wo = W >> 1;
  int idx = blockIdx.x * blockDim.x + threadIdx.x;
  int total = C * Ho * Wo;
  if (idx >= total) return;
  int x = idx % Wo, y = (idx / Wo) % Ho, c = idx / (Wo * Ho);
  const float* p = in + ((size_t)c * H + 2 * y) * W + 2 * x;
  out[idx] = fmaxf(fmaxf(p[0], p[1]), fmaxf(p[W], p[W + 1]));
}

// ---------------------------------------------------------------------------
// 1x1 conv, output written pixel-major (== reference transpose+reshape flat)
// ---------------------------------------------------------------------------
__global__ void conv1x1_kernel(const float* __restrict__ in,
                               const float* __restrict__ w,
                               const float* __restrict__ b,
                               float* __restrict__ out,
                               int Cin, int Cout, int P)
{
  int idx = blockIdx.x * blockDim.x + threadIdx.x;
  if (idx >= P * Cout) return;
  int co = idx % Cout, p = idx / Cout;
  float s = b[co];
  for (int ci = 0; ci < Cin; ++ci)
    s += in[(size_t)ci * P + p] * w[(size_t)co * Cin + ci];
  out[(size_t)p * Cout + co] = s;
}

// ---------------------------------------------------------------------------
// Adaptive max ROI pooling to 7x7 (matches reference incl. degenerate cases)
// ---------------------------------------------------------------------------
__global__ void roi_pool_kernel(const float* __restrict__ feat,
                                const float* __restrict__ rois,
                                const int* __restrict__ roi_index,
                                float* __restrict__ pooled,
                                int R, int C, int H, int W)
{
  int idx = blockIdx.x * blockDim.x + threadIdx.x;
  int total = R * C * 49;
  if (idx >= total) return;
  int ox = idx % 7, oy = (idx / 7) % 7, c = (idx / 49) % C, r = idx / (49 * C);

  const float scale = 1.0f / 16.0f;
  int x1 = (int)rintf(rois[r * 4 + 0] * scale);
  int y1 = (int)rintf(rois[r * 4 + 1] * scale);
  int x2 = (int)rintf(rois[r * 4 + 2] * scale);
  int y2 = (int)rintf(rois[r * 4 + 3] * scale);
  x1 = min(max(x1, 0), W); y1 = min(max(y1, 0), H);
  x2 = min(max(x2, 0), W); y2 = min(max(y2, 0), H);
  bool valid = (x1 < x2) && (y1 < y2);
  int h = y2 - y1 + 1, w = x2 - x1 + 1;

  int ys = y1 + (oy * h) / 7;
  int ye = y1 + ((oy + 1) * h + 6) / 7;
  int xs = x1 + (ox * w) / 7;
  int xe = x1 + ((ox + 1) * w + 6) / 7;
  ye = min(ye, H); xe = min(xe, W);

  float m = -3.402823466e+38f;  // finfo(float32).min
  const float* base = feat + ((size_t)roi_index[r] * C + c) * H * W;
  for (int y = ys; y < ye; ++y)
    for (int x = xs; x < xe; ++x)
      m = fmaxf(m, base[(size_t)y * W + x]);
  if (!valid) m = 0.0f;
  pooled[idx] = m;
}

// ---------------------------------------------------------------------------
// Host-side orchestration (graph-capture safe: only stream launches)
// ---------------------------------------------------------------------------
extern "C" void kernel_launch(void* const* d_in, const int* in_sizes, int n_in,
                              void* d_out, int out_size, void* d_ws, size_t ws_size,
                              hipStream_t stream)
{
  (void)in_sizes; (void)n_in; (void)out_size; (void)ws_size;

  const float* x         = (const float*)d_in[0];
  const float* rois      = (const float*)d_in[1];
  const int*   roi_index = (const int*)d_in[2];
  auto P_ = [&](int i) { return (const float*)d_in[i]; };
  // params flat order: vgg w/b x13 (3..28), rpn_conv (29,30), rpn_score (31,32),
  // rpn_loc (33,34), fc1 (35,36), fc2 (37,38), head_loc (39,40), head_score (41,42)

  float* ws  = (float*)d_ws;
  float* A   = ws;                   // 64*600*800 = 30,720,000 floats
  float* B   = A + 30720000;         // ping-pong
  float* P0  = B + 30720000;         // pooled: 128*25088
  float* F1  = P0 + 3211264;         // 128*4096
  float* F2  = F1 + 524288;          // 128*4096
  __bf16* WRE = (__bf16*)(F2 + 524288);  // repacked weights: <= 9*512*512 bf16
  float* out = (float*)d_out;

  auto conv = [&](const float* in, int wi, float* o,
                  int Cin, int Cout, int H, int W, int relu) {
    int CinPad = (Cin + 31) & ~31;
    int rtot = 9 * Cout * CinPad;
    repack_w_kernel<<<(rtot + 255) / 256, 256, 0, stream>>>(
        P_(wi), WRE, Cin, CinPad, Cout);
    dim3 g((W + 31) / 32, H, Cout / 16);
    conv3x3_wmma<<<g, 32, 0, stream>>>(in, WRE, P_(wi + 1), o,
                                       Cin, CinPad, Cout, H, W, relu);
  };
  auto pool = [&](const float* in, float* o, int C, int H, int W) {
    int tot = C * (H / 2) * (W / 2);
    maxpool2_kernel<<<(tot + 255) / 256, 256, 0, stream>>>(in, o, C, H, W);
  };

  // VGG16 trunk (features[:30], last pool dropped)
  conv(x, 3,  A, 3,   64, 600, 800, 1);
  conv(A, 5,  B, 64,  64, 600, 800, 1);
  pool(B, A, 64, 600, 800);                       // 64 x300x400
  conv(A, 7,  B, 64, 128, 300, 400, 1);
  conv(B, 9,  A, 128,128, 300, 400, 1);
  pool(A, B, 128, 300, 400);                      // 128x150x200
  conv(B, 11, A, 128,256, 150, 200, 1);
  conv(A, 13, B, 256,256, 150, 200, 1);
  conv(B, 15, A, 256,256, 150, 200, 1);
  pool(A, B, 256, 150, 200);                      // 256x 75x100
  conv(B, 17, A, 256,512,  75, 100, 1);
  conv(A, 19, B, 512,512,  75, 100, 1);
  conv(B, 21, A, 512,512,  75, 100, 1);
  pool(A, B, 512, 75, 100);                       // 512x 37x 50
  conv(B, 23, A, 512,512,  37,  50, 1);
  conv(A, 25, B, 512,512,  37,  50, 1);
  conv(B, 27, A, 512,512,  37,  50, 1);           // feat -> A

  // Output layout: [cls_loc 128*4008 | scores 128*1002 | rpn_loc 66600 | rpn_score 33300]
  const int OFF_SCORE  = 128 * 4008;              // 513024
  const int OFF_RPNLOC = OFF_SCORE + 128 * 1002;  // 641280
  const int OFF_RPNSCR = OFF_RPNLOC + 37 * 50 * 36;

  // RPN
  conv(A, 29, B, 512, 512, 37, 50, 1);            // mid -> B
  const int P = 37 * 50;
  conv1x1_kernel<<<(P * 36 + 255) / 256, 256, 0, stream>>>(
      B, P_(33), P_(34), out + OFF_RPNLOC, 512, 36, P);
  conv1x1_kernel<<<(P * 18 + 255) / 256, 256, 0, stream>>>(
      B, P_(31), P_(32), out + OFF_RPNSCR, 512, 18, P);

  // ROI pooling: feat(A) -> P0 [128, 512*7*7]
  {
    int tot = 128 * 512 * 49;
    roi_pool_kernel<<<(tot + 255) / 256, 256, 0, stream>>>(
        A, rois, roi_index, P0, 128, 512, 37, 50);
  }

  // FC head
  { dim3 g(4096 / 16, 8);
    gemm_wmma<<<g, 32, 0, stream>>>(P0, P_(35), P_(36), F1, 128, 4096, 25088, 1); }
  { dim3 g(4096 / 16, 8);
    gemm_wmma<<<g, 32, 0, stream>>>(F1, P_(37), P_(38), F2, 128, 4096, 4096, 1); }
  { dim3 g((4008 + 15) / 16, 8);
    gemm_wmma<<<g, 32, 0, stream>>>(F2, P_(39), P_(40), out, 128, 4008, 4096, 0); }
  { dim3 g((1002 + 15) / 16, 8);
    gemm_wmma<<<g, 32, 0, stream>>>(F2, P_(41), P_(42), out + OFF_SCORE, 128, 1002, 4096, 0); }
}
